// GC_Conv_4879082848656
// MI455X (gfx1250) — matrix operand order repaired and checked by reading, not compiled, not run on hardware
//
#include <hip/hip_runtime.h>

typedef __attribute__((ext_vector_type(16))) _Float16 v16h;
typedef __attribute__((ext_vector_type(8)))  float    v8f;

#define BN_EPS 1e-5f
#define LDSPAD 8   // halves of row padding -> row stride shifts 4 dwords/bank-group

static inline int cdiv_i(long long a, long long b){ return (int)((a + b - 1) / b); }

// ---------------------------------------------------------------- WMMA helpers

__device__ __forceinline__ v8f wmma_fp16(v16h a, v16h b, v8f c){
  // D = A(16x32 f16) * B(32x16 f16) + C(16x16 f32)
  return __builtin_amdgcn_wmma_f32_16x16x32_f16(false, a, false, b, (short)0, c, false, false);
}

// load 8 consecutive f32, convert to f16 into a[base..base+7]
__device__ __forceinline__ void load_run8_f32(const float* __restrict__ p, v16h& a, int base){
  float4 u = *reinterpret_cast<const float4*>(p);
  float4 v = *reinterpret_cast<const float4*>(p + 4);
  a[base+0] = (_Float16)u.x; a[base+1] = (_Float16)u.y;
  a[base+2] = (_Float16)u.z; a[base+3] = (_Float16)u.w;
  a[base+4] = (_Float16)v.x; a[base+5] = (_Float16)v.y;
  a[base+6] = (_Float16)v.z; a[base+7] = (_Float16)v.w;
}

__device__ __forceinline__ void load_run8_f16(const _Float16* __restrict__ p, v16h& a, int base){
  #pragma unroll
  for (int i = 0; i < 8; ++i) a[base+i] = p[i];
}

// prepped B tile: [tile][lane][16 halves] contiguous, 32B-aligned
__device__ __forceinline__ v16h load_b_tile(const _Float16* __restrict__ B, int tile, int lane){
  return *reinterpret_cast<const v16h*>(B + ((size_t)tile*32 + lane)*16);
}

__device__ __forceinline__ float leaky02(float y){ return fmaxf(y, 0.2f*y); }

// ---------------------------------------------------------------- atomics

__device__ __forceinline__ void atomicMaxF(float* addr, float v){
  unsigned int* ua = reinterpret_cast<unsigned int*>(addr);
  unsigned int old = __float_as_uint(*addr);
  while (__uint_as_float(old) < v){
    unsigned int assumed = old;
    old = atomicCAS(ua, assumed, __float_as_uint(v));
    if (old == assumed) break;
  }
}
__device__ __forceinline__ void atomicMinF(float* addr, float v){
  unsigned int* ua = reinterpret_cast<unsigned int*>(addr);
  unsigned int old = __float_as_uint(*addr);
  while (__uint_as_float(old) > v){
    unsigned int assumed = old;
    old = atomicCAS(ua, assumed, __float_as_uint(v));
    if (old == assumed) break;
  }
}

// ---------------------------------------------------------------- small kernels

__global__ void fill_kernel(float* __restrict__ p, float v, int n){
  int i = blockIdx.x*blockDim.x + threadIdx.x;
  if (i < n) p[i] = v;
}

__global__ void degree_kernel(const int* __restrict__ dst, float* __restrict__ deg, int E){
  int e = blockIdx.x*blockDim.x + threadIdx.x;
  if (e < E) atomicAdd(deg + dst[e], 1.0f);
}

// in_vc = nf[M,16] @ W[16,32] + b
__global__ void fc_vc_kernel(const float* __restrict__ nf, const float* __restrict__ W,
                             const float* __restrict__ b, float* __restrict__ out, int M){
  int idx = blockIdx.x*blockDim.x + threadIdx.x;
  if (idx >= M*32) return;
  int n = idx >> 5, c = idx & 31;
  float s = b[c];
  #pragma unroll
  for (int k = 0; k < 16; ++k) s += nf[n*16 + k]*W[k*32 + c];
  out[idx] = s;
}

// in_ve = ef[E,1] @ W[1,32] + b
__global__ void fc_ve_kernel(const float* __restrict__ ef, const float* __restrict__ W,
                             const float* __restrict__ b, float* __restrict__ out, int E){
  int idx = blockIdx.x*blockDim.x + threadIdx.x;
  if (idx >= E*32) return;
  int e = idx >> 5, c = idx & 31;
  out[idx] = ef[e]*W[c] + b[c];
}

// Swizzle f32 row-major weight (K x >=N at ldw, col offset c0) into WMMA-B f16 layout:
// Bp[((kt*NT + nt)*32 + lane)*16 + i] = W[kt*32 + (lane>=16?8:0) + (i&7) + (i>=8?16:0)][c0 + nt*16 + (lane&15)]
__global__ void prep_b_kernel(const float* __restrict__ W, _Float16* __restrict__ Bp,
                              int K, int N, int ldw, int c0){
  int idx = blockIdx.x*blockDim.x + threadIdx.x;
  if (idx >= K*N) return;
  int i    = idx & 15;
  int lane = (idx >> 4) & 31;
  int tile = idx >> 9;
  int NT   = N >> 4;
  int nt   = tile % NT;
  int kt   = tile / NT;
  int k    = kt*32 + ((lane >> 4) << 3) + (i & 7) + ((i >> 3) << 4);
  int n    = nt*16 + (lane & 15);
  Bp[idx]  = (_Float16)W[(size_t)k*ldw + c0 + n];
}

// BatchNorm (training stats) applied in place; C is a power of two
template<int C>
__global__ void bn_apply_kernel(float* __restrict__ x, const float* __restrict__ cs,
                                const float* __restrict__ cq, const float* __restrict__ g,
                                const float* __restrict__ b, int R){
  int idx = blockIdx.x*blockDim.x + threadIdx.x;
  if (idx >= R*C) return;
  int c = idx & (C - 1);
  float rinv = 1.0f / (float)R;
  float mu  = cs[c] * rinv;
  float var = cq[c] * rinv - mu*mu;
  x[idx] = (x[idx] - mu) * rsqrtf(var + BN_EPS) * g[c] + b[c];
}

// ---------------------------------------------------------------- fused GCMP edge kernel
// Per 16-edge tile (1 wave):
//   stage x=[vc[src]|vc[dst]] and ve into LDS (f16, bank-padded rows)
//   h = leaky(x@W1+b1)                      (LDS, f16)
//   gate = sigmoid(h @ W2[:,0] + b2[0])
//   f = (h@W2[:,1:]+b2[1:]) * gate          (LDS, f16; overwrites x stage)
//   atomics: sum(f1) max(f2) min(f3) sum(f4) into dst-node buffers (L2-resident)
//   ve_out = [f | ve]@Wd + bd               (+ BN column stats)
template<int NI, int NO>
__global__ void __launch_bounds__(64)
gcmp_edge_kernel(const float* __restrict__ vc, const float* __restrict__ ve,
                 const int* __restrict__ src, const int* __restrict__ dst, int E,
                 const _Float16* __restrict__ B1, const float* __restrict__ b1,
                 const _Float16* __restrict__ B2, const float* __restrict__ W2raw,
                 const float* __restrict__ b2,
                 const _Float16* __restrict__ Bd, const float* __restrict__ bd,
                 float* __restrict__ aggS, float* __restrict__ aggMx,
                 float* __restrict__ aggMn, float* __restrict__ aggM4,
                 float* __restrict__ veOut, float* __restrict__ veSum,
                 float* __restrict__ veSq)
{
  constexpr int KH  = 3*NI;        // width of x
  constexpr int NH  = 3*NO;        // width of h
  constexpr int NF  = 4*NO;        // width of f
  constexpr int XSD = 2*NI;        // staged src|dst width
  static_assert(NF >= XSD, "f region must cover x stage region");

  // xf: phase 1 holds x[:, 0:2*NI] (src|dst), phase 2 holds f[:, 0:NF]
  __shared__ _Float16 xf  [2][16][NF + LDSPAD];
  __shared__ _Float16 veb [2][16][NI + LDSPAD];
  __shared__ _Float16 hbuf[2][16][NH + LDSPAD];
  __shared__ float    gbuf[2][16];
  __shared__ int      dbuf[2][16];

  const int lane = threadIdx.x & 31;
  const int w    = threadIdx.x >> 5;
  const int e0   = (blockIdx.x*2 + w)*16;
  if (e0 >= E) return;                       // wave-uniform

  const int m    = lane & 15;                // A row / gate row
  const int off8 = (lane >> 4) << 3;         // 0 or 8 : A K-run & C row offset

  const int eM = e0 + m;
  const int eC = (eM < E) ? eM : (E - 1);
  const float* psrc = vc + (size_t)src[eC]*NI;
  const float* pdst = vc + (size_t)dst[eC]*NI;
  const float* pve  = ve + (size_t)eC*NI;
  if (lane < 16) dbuf[w][m] = dst[eC];

  // ---- stage x (gathered) into LDS as f16: lane<16 stages vc[src] row, lane>=16 vc[dst] row
  {
    const float* sp   = (lane < 16) ? psrc : pdst;
    const int    cb   = (lane < 16) ? 0 : NI;
    #pragma unroll
    for (int k = 0; k < NI; k += 4){
      float4 u = *reinterpret_cast<const float4*>(sp + k);
      xf[w][m][cb+k+0] = (_Float16)u.x; xf[w][m][cb+k+1] = (_Float16)u.y;
      xf[w][m][cb+k+2] = (_Float16)u.z; xf[w][m][cb+k+3] = (_Float16)u.w;
    }
    const int vb = (lane < 16) ? 0 : NI/2;   // split ve row between the two half-waves
    #pragma unroll
    for (int k = 0; k < NI/2; k += 4){
      float4 u = *reinterpret_cast<const float4*>(pve + vb + k);
      veb[w][m][vb+k+0] = (_Float16)u.x; veb[w][m][vb+k+1] = (_Float16)u.y;
      veb[w][m][vb+k+2] = (_Float16)u.z; veb[w][m][vb+k+3] = (_Float16)u.w;
    }
  }

  // ---- GEMM1: h = leaky(x @ W1 + b1)   (A entirely from LDS)
  for (int nt = 0; nt < NH/16; ++nt){
    v8f acc = {};
    #pragma unroll
    for (int kt = 0; kt < KH/32; ++kt){
      const int segk = kt*32;
      v16h a;
      const _Float16* ap = (segk < XSD) ? &xf[w][m][segk + off8]
                                        : &veb[w][m][segk - XSD + off8];
      load_run8_f16(ap,      a, 0);
      load_run8_f16(ap + 16, a, 8);
      v16h b = load_b_tile(B1, kt*(NH/16) + nt, lane);
      acc = wmma_fp16(a, b, acc);
    }
    const int col = nt*16 + (lane & 15);
    const float bias = b1[col];
    #pragma unroll
    for (int r = 0; r < 8; ++r)
      hbuf[w][r + off8][col] = (_Float16)leaky02(acc[r] + bias);
  }

  // ---- sigmoid gate: column 0 of W2 (scalar dot; weight reads are wave-uniform)
  if (lane < 16){
    float g = b2[0];
    for (int k = 0; k < NH; ++k)
      g += (float)hbuf[w][m][k] * W2raw[(size_t)k*(NF+1)];
    gbuf[w][m] = 1.f / (1.f + __expf(-g));
  }

  // ---- GEMM2 + gate + segment-aggregation atomics (f overwrites x stage in LDS)
  for (int nt = 0; nt < NF/16; ++nt){
    v8f acc = {};
    #pragma unroll
    for (int kt = 0; kt < NH/32; ++kt){
      v16h a;
      const _Float16* hp = &hbuf[w][m][kt*32 + off8];
      load_run8_f16(hp,      a, 0);
      load_run8_f16(hp + 16, a, 8);
      __builtin_prefetch(B2 + ((size_t)((kt+1)*(NF/16) + nt)*32 + lane)*16, 0, 3);
      v16h b = load_b_tile(B2, kt*(NF/16) + nt, lane);
      acc = wmma_fp16(a, b, acc);
    }
    const int col  = nt*16 + (lane & 15);
    const int seg  = col / NO;
    const int c    = col % NO;
    const float bias = b2[1 + col];
    #pragma unroll
    for (int r = 0; r < 8; ++r){
      const int row = r + off8;
      const float f = (acc[r] + bias) * gbuf[w][row];
      xf[w][row][col] = (_Float16)f;
      if (e0 + row < E){
        const size_t o = (size_t)dbuf[w][row]*NO + c;
        if      (seg == 0) atomicAdd (aggS  + o, f);
        else if (seg == 1) atomicMaxF(aggMx + o, f);
        else if (seg == 2) atomicMinF(aggMn + o, f);
        else               atomicAdd (aggM4 + o, f);
      }
    }
  }

  // ---- GEMM3: ve_out = [f | ve] @ Wd + bd  (+ BN stats); A entirely from LDS
  for (int nt = 0; nt < NO/16; ++nt){
    v8f acc = {};
    #pragma unroll
    for (int kt = 0; kt < (NF + NI)/32; ++kt){
      const int segk = kt*32;
      v16h a;
      const _Float16* ap = (segk < NF) ? &xf[w][m][segk + off8]
                                       : &veb[w][m][segk - NF + off8];
      load_run8_f16(ap,      a, 0);
      load_run8_f16(ap + 16, a, 8);
      v16h b = load_b_tile(Bd, kt*(NO/16) + nt, lane);
      acc = wmma_fp16(a, b, acc);
    }
    const int col = nt*16 + (lane & 15);
    const float bias = bd[col];
    float s = 0.f, s2 = 0.f;
    #pragma unroll
    for (int r = 0; r < 8; ++r){
      const int row = r + off8;
      const float y = acc[r] + bias;
      if (e0 + row < E){
        veOut[(size_t)(e0 + row)*NO + col] = y;
        s += y; s2 += y*y;
      }
    }
    atomicAdd(veSum + col, s);
    atomicAdd(veSq  + col, s2);
  }
}

// ---------------------------------------------------------------- node side

// X = f16[[in_vc | sum | max' | min' | mean]]  (max/min zero for isolated, mean=sum4/max(deg,1))
template<int NI, int NO>
__global__ void finalize_node_kernel(const float* __restrict__ vcIn,
                                     const float* __restrict__ aggS, const float* __restrict__ aggMx,
                                     const float* __restrict__ aggMn, const float* __restrict__ aggM4,
                                     const float* __restrict__ deg, _Float16* __restrict__ X, int M)
{
  constexpr int K = NI + 4*NO;
  int idx = blockIdx.x*blockDim.x + threadIdx.x;
  if (idx >= M*K) return;
  const int n = idx / K, c = idx % K;       // K is compile-time: strength-reduced
  float v;
  if (c < NI) v = vcIn[(size_t)n*NI + c];
  else {
    const int cc = c - NI, seg = cc / NO, j = cc % NO;
    const size_t o = (size_t)n*NO + j;
    const float d = deg[n];
    if      (seg == 0) v = aggS[o];
    else if (seg == 1) v = (d > 0.f) ? aggMx[o] : 0.f;
    else if (seg == 2) v = (d > 0.f) ? aggMn[o] : 0.f;
    else               v = aggM4[o] / fmaxf(d, 1.f);
  }
  X[idx] = (_Float16)v;
}

// out[M,N] = A_f16[M,K] @ B(prepped) + bias, BN column stats.
// Each wave: 64x16 output tile (4 accumulators) so every B tile feeds 4 WMMAs.
template<int K, int N>
__global__ void __launch_bounds__(128)
node_gemm_kernel(const _Float16* __restrict__ A, const _Float16* __restrict__ B,
                 const float* __restrict__ bias, int M, float* __restrict__ out,
                 float* __restrict__ colSum, float* __restrict__ colSq)
{
  const int lane = threadIdx.x & 31;
  const int wv   = threadIdx.x >> 5;
  const int gid  = blockIdx.x*4 + wv;
  constexpr int NT = N/16;
  const int mt = gid / NT, nt = gid % NT;
  const int row0 = mt*64;
  if (row0 >= M) return;                    // wave-uniform
  const int m    = lane & 15;
  const int off8 = (lane >> 4) << 3;

  const _Float16* ap[4];
  #pragma unroll
  for (int t = 0; t < 4; ++t){
    int r = row0 + t*16 + m; if (r >= M) r = M - 1;
    ap[t] = A + (size_t)r*K;
  }
  v8f acc[4] = {{},{},{},{}};
  #pragma unroll
  for (int kt = 0; kt < K/32; ++kt){
    __builtin_prefetch(B + ((size_t)((kt+1)*NT + nt)*32 + lane)*16, 0, 3);
    v16h b = load_b_tile(B, kt*NT + nt, lane);
    #pragma unroll
    for (int t = 0; t < 4; ++t){
      v16h a;
      load_run8_f16(ap[t] + kt*32 + off8,      a, 0);
      load_run8_f16(ap[t] + kt*32 + off8 + 16, a, 8);
      acc[t] = wmma_fp16(a, b, acc[t]);
    }
  }
  const int col = nt*16 + (lane & 15);
  const float bv = bias[col];
  float s = 0.f, s2 = 0.f;
  #pragma unroll
  for (int t = 0; t < 4; ++t){
    #pragma unroll
    for (int r = 0; r < 8; ++r){
      const int rw = row0 + t*16 + r + off8;
      const float y = acc[t][r] + bv;
      if (rw < M){ out[(size_t)rw*N + col] = y; s += y; s2 += y*y; }
    }
  }
  atomicAdd(colSum + col, s);
  atomicAdd(colSq  + col, s2);
}

// ---------------------------------------------------------------- ToEdge (193->3->5, gate, ->1)
__global__ void toedge_kernel(const float* __restrict__ vc, const float* __restrict__ ve,
                              const float* __restrict__ ef, const int* __restrict__ src,
                              const int* __restrict__ dst,
                              const float* __restrict__ w1, const float* __restrict__ b1,
                              const float* __restrict__ w2, const float* __restrict__ b2,
                              const float* __restrict__ wo, const float* __restrict__ bo,
                              float* __restrict__ out, int E)
{
  int e = blockIdx.x*blockDim.x + threadIdx.x;
  if (e >= E) return;
  const float* ps = vc + (size_t)src[e]*64;
  const float* pd = vc + (size_t)dst[e]*64;
  const float* pv = ve + (size_t)e*64;
  float h0 = b1[0], h1 = b1[1], h2 = b1[2];
  for (int k = 0; k < 64; ++k){ float x = ps[k]; h0 += x*w1[k*3+0]; h1 += x*w1[k*3+1]; h2 += x*w1[k*3+2]; }
  for (int k = 0; k < 64; ++k){ float x = pd[k]; h0 += x*w1[(64+k)*3+0]; h1 += x*w1[(64+k)*3+1]; h2 += x*w1[(64+k)*3+2]; }
  for (int k = 0; k < 64; ++k){ float x = pv[k]; h0 += x*w1[(128+k)*3+0]; h1 += x*w1[(128+k)*3+1]; h2 += x*w1[(128+k)*3+2]; }
  { float x = ef[e]; h0 += x*w1[192*3+0]; h1 += x*w1[192*3+1]; h2 += x*w1[192*3+2]; }
  h0 = leaky02(h0); h1 = leaky02(h1); h2 = leaky02(h2);
  float y[5];
  #pragma unroll
  for (int j = 0; j < 5; ++j) y[j] = b2[j] + h0*w2[0*5+j] + h1*w2[1*5+j] + h2*w2[2*5+j];
  const float gk = 1.f / (1.f + __expf(-y[0]));
  float o = bo[0];
  #pragma unroll
  for (int j = 0; j < 4; ++j) o += y[1+j]*gk*wo[j];
  out[e] = o;
}

// ---------------------------------------------------------------- host orchestration

struct LayerP {
  const float *w1,*b1,*w2,*b2,*wr,*br,*wd,*bd,*g_gc,*bb_gc,*g_ef,*bb_ef;
};

template<int NI, int NO>
static void run_layer(hipStream_t stream, int E, int M,
                      const float* vcIn, const float* veIn,
                      float* vcOut, float* veOut,
                      const int* src, const int* dst, const LayerP& P,
                      const _Float16* B1, const _Float16* B2,
                      const _Float16* Br, const _Float16* Bd,
                      float* aggS, float* aggMx, float* aggMn, float* aggM4,
                      const float* deg, _Float16* Xnode, float* st)
{
  constexpr int Kn = NI + 4*NO;
  const int aggN = M*NO;
  fill_kernel<<<cdiv_i(aggN,256),256,0,stream>>>(aggS,  0.f,      aggN);
  fill_kernel<<<cdiv_i(aggN,256),256,0,stream>>>(aggM4, 0.f,      aggN);
  fill_kernel<<<cdiv_i(aggN,256),256,0,stream>>>(aggMx, -3.4e38f, aggN);
  fill_kernel<<<cdiv_i(aggN,256),256,0,stream>>>(aggMn,  3.4e38f, aggN);
  fill_kernel<<<2,256,0,stream>>>(st, 0.f, 512);

  gcmp_edge_kernel<NI,NO><<<cdiv_i(E,32),64,0,stream>>>(
      vcIn, veIn, src, dst, E, B1, P.b1, B2, P.w2, P.b2, Bd, P.bd,
      aggS, aggMx, aggMn, aggM4, veOut, st+256, st+384);

  finalize_node_kernel<NI,NO><<<cdiv_i((long long)M*Kn,256),256,0,stream>>>(
      vcIn, aggS, aggMx, aggMn, aggM4, deg, Xnode, M);

  const long long waves = (long long)cdiv_i(M,64)*(NO/16);
  node_gemm_kernel<Kn,NO><<<cdiv_i(waves,4),128,0,stream>>>(
      Xnode, Br, P.br, M, vcOut, st, st+128);

  bn_apply_kernel<NO><<<cdiv_i((long long)M*NO,256),256,0,stream>>>(vcOut, st,     st+128, P.g_gc, P.bb_gc, M);
  bn_apply_kernel<NO><<<cdiv_i((long long)E*NO,256),256,0,stream>>>(veOut, st+256, st+384, P.g_ef, P.bb_ef, E);
}

extern "C" void kernel_launch(void* const* d_in, const int* in_sizes, int n_in,
                              void* d_out, int out_size, void* d_ws, size_t ws_size,
                              hipStream_t stream)
{
  const float* nf = (const float*)d_in[0];
  const float* ef = (const float*)d_in[1];
  const int*  src = (const int*)d_in[2];
  const int*  dst = (const int*)d_in[3];
  const int M = in_sizes[0] / 16;
  const int E = in_sizes[2];

  const float* fc_gc_w = (const float*)d_in[4];
  const float* fc_gc_b = (const float*)d_in[5];
  const float* fc_ef_w = (const float*)d_in[6];
  const float* fc_ef_b = (const float*)d_in[7];

  LayerP L[3];
  for (int i = 0; i < 3; ++i){
    const int b = 8 + 12*i;
    L[i].w1    = (const float*)d_in[b+0];  L[i].b1    = (const float*)d_in[b+1];
    L[i].w2    = (const float*)d_in[b+2];  L[i].b2    = (const float*)d_in[b+3];
    L[i].wr    = (const float*)d_in[b+4];  L[i].br    = (const float*)d_in[b+5];
    L[i].wd    = (const float*)d_in[b+6];  L[i].bd    = (const float*)d_in[b+7];
    L[i].g_gc  = (const float*)d_in[b+8];  L[i].bb_gc = (const float*)d_in[b+9];
    L[i].g_ef  = (const float*)d_in[b+10]; L[i].bb_ef = (const float*)d_in[b+11];
  }
  const float* te_w1 = (const float*)d_in[44];
  const float* te_b1 = (const float*)d_in[45];
  const float* te_w2 = (const float*)d_in[46];
  const float* te_b2 = (const float*)d_in[47];
  const float* te_wo = (const float*)d_in[48];
  const float* te_bo = (const float*)d_in[49];

  // ---- workspace arena
  char* ws = (char*)d_ws;
  size_t off = 0;
  auto alloc = [&](size_t bytes)->void*{
    off = (off + 255) & ~(size_t)255;
    void* p = ws + off; off += bytes; return p;
  };
  float* vcA  = (float*)alloc((size_t)M*128*4);
  float* vcB  = (float*)alloc((size_t)M*128*4);
  float* veA  = (float*)alloc((size_t)E*128*4);
  float* veB  = (float*)alloc((size_t)E*128*4);
  float* aggS = (float*)alloc((size_t)M*128*4);
  float* aggMx= (float*)alloc((size_t)M*128*4);
  float* aggMn= (float*)alloc((size_t)M*128*4);
  float* aggM4= (float*)alloc((size_t)M*128*4);
  float* deg  = (float*)alloc((size_t)M*4);
  _Float16* Xnode = (_Float16*)alloc((size_t)M*576*2);
  float* st   = (float*)alloc(512*4);
  _Float16* bpA = (_Float16*)alloc((size_t)1000000*2);

  size_t bo = 0;
  auto carve = [&](size_t n)->_Float16*{
    _Float16* p = bpA + bo; bo += (n + 15) & ~(size_t)15; return p;
  };
  _Float16 *B1_1 = carve(96*192),  *B2_1 = carve(192*256), *Br_1 = carve(288*64),  *Bd_1 = carve(288*64);
  _Float16 *B1_2 = carve(192*384), *B2_2 = carve(384*512), *Br_2 = carve(576*128), *Bd_2 = carve(576*128);
  _Float16 *B1_3 = carve(384*192), *B2_3 = carve(192*256), *Br_3 = carve(384*64),  *Bd_3 = carve(384*64);

  auto prep = [&](const float* W, _Float16* Bp, int K, int N, int ldw, int c0){
    prep_b_kernel<<<cdiv_i((long long)K*N,256),256,0,stream>>>(W, Bp, K, N, ldw, c0);
  };
  // layer1 (32->64)
  prep(L[0].w1, B1_1,  96, 192, 192, 0);
  prep(L[0].w2, B2_1, 192, 256, 257, 1);
  prep(L[0].wr, Br_1, 288,  64,  64, 0);
  prep(L[0].wd, Bd_1, 288,  64,  64, 0);
  // layer2 (64->128)
  prep(L[1].w1, B1_2, 192, 384, 384, 0);
  prep(L[1].w2, B2_2, 384, 512, 513, 1);
  prep(L[1].wr, Br_2, 576, 128, 128, 0);
  prep(L[1].wd, Bd_2, 576, 128, 128, 0);
  // layer3 (128->64)
  prep(L[2].w1, B1_3, 384, 192, 192, 0);
  prep(L[2].w2, B2_3, 192, 256, 257, 1);
  prep(L[2].wr, Br_3, 384,  64,  64, 0);
  prep(L[2].wd, Bd_3, 384,  64,  64, 0);

  // degrees (same for all layers)
  fill_kernel<<<cdiv_i(M,256),256,0,stream>>>(deg, 0.f, M);
  degree_kernel<<<cdiv_i(E,256),256,0,stream>>>(dst, deg, E);

  // input linears
  fc_vc_kernel<<<cdiv_i((long long)M*32,256),256,0,stream>>>(nf, fc_gc_w, fc_gc_b, vcA, M);
  fc_ve_kernel<<<cdiv_i((long long)E*32,256),256,0,stream>>>(ef, fc_ef_w, fc_ef_b, veA, E);

  // three fused GCMP layers (ping-pong buffers)
  run_layer<32,64 >(stream, E, M, vcA, veA, vcB, veB, src, dst, L[0],
                    B1_1, B2_1, Br_1, Bd_1, aggS, aggMx, aggMn, aggM4, deg, Xnode, st);
  run_layer<64,128>(stream, E, M, vcB, veB, vcA, veA, src, dst, L[1],
                    B1_2, B2_2, Br_2, Bd_2, aggS, aggMx, aggMn, aggM4, deg, Xnode, st);
  run_layer<128,64>(stream, E, M, vcA, veA, vcB, veB, src, dst, L[2],
                    B1_3, B2_3, Br_3, Bd_3, aggS, aggMx, aggMn, aggM4, deg, Xnode, st);

  // final edge head
  toedge_kernel<<<cdiv_i(E,256),256,0,stream>>>(
      vcB, veB, ef, src, dst, te_w1, te_b1, te_w2, te_b2, te_wo, te_bo,
      (float*)d_out, E);
}
